// times_block_52183852646547
// MI455X (gfx1250) — matrix operand order repaired and checked by reading, not compiled, not run on hardware
//
#include <hip/hip_runtime.h>
#include <hip/hip_bf16.h>

#define BB 8
#define TT 4096
#define CC 256
#define KK 4
#define FB 2048   // T/2 frequency bins (DC dropped)

typedef __attribute__((ext_vector_type(16))) __bf16 v16bf;
typedef __attribute__((ext_vector_type(8)))  float  v8f;

union Frag {
  v16bf v;
  uint4 q[2];
};

__device__ __forceinline__ unsigned short f2bf(float f) {
  union { float f; unsigned u; } u;
  u.f = f;
  unsigned r = u.u + 0x7FFFu + ((u.u >> 16) & 1u);   // round-to-nearest-even
  return (unsigned short)(r >> 16);
}
__device__ __forceinline__ float bf2f(unsigned short h) {
  union { unsigned u; float f; } u;
  u.u = ((unsigned)h) << 16;
  return u.f;
}
__device__ __forceinline__ float gelu_exact(float v) {
  return 0.5f * v * (1.0f + erff(v * 0.70710678118654752f));
}

// ---------------------------------------------------------------- x -> bf16
__global__ void k_xbf(const float* __restrict__ x, unsigned short* __restrict__ xbf, int n) {
  int i = blockIdx.x * 256 + threadIdx.x;
  if (i < n) xbf[i] = f2bf(x[i]);
}

// ------------------------------------------- weights -> WMMA fragment layout
// For a K x N weight (row-major), B-fragment of tile (kc, nt):
//   lane L: hl=L/16, nl=L%16 ; element i (0..15): k = kc*32 + 16*hl + i, n = nt*16 + nl
// stored contiguously: dst[((m*(K/32)+kc)*(N/16)+nt)*512 + L*16 + i]  (bf16)
__global__ void k_frags(const float* __restrict__ src, unsigned short* __restrict__ dst,
                        int K, int N, int nMat) {
  int total = nMat * K * N;
  int e = blockIdx.x * 256 + threadIdx.x;
  if (e >= total) return;
  int i    = e & 15;
  int lane = (e >> 4) & 31;
  int tIdx = e >> 9;
  int NT = N >> 4, KC = K >> 5;
  int nt = tIdx % NT;
  int kc = (tIdx / NT) % KC;
  int m  = tIdx / (NT * KC);
  int hl = lane >> 4, nl = lane & 15;
  int kk = kc * 32 + 16 * hl + i;
  int nn = nt * 16 + nl;
  dst[e] = f2bf(src[((size_t)m * K + kk) * N + nn]);
}

// --------------------------------------- 4096-pt radix-2 FFT per (b,c) in LDS
__global__ __launch_bounds__(256) void k_fft(const float* __restrict__ x,
                                             float* __restrict__ mag) {
  __shared__ float re[TT];
  __shared__ float im[TT];
  int b = blockIdx.x >> 8, c = blockIdx.x & 255;
  int tid = threadIdx.x;
  const float* xp = x + ((size_t)b * TT) * CC + c;
  for (int t = tid; t < TT; t += 256) {          // bit-reversed load
    int r = __brev((unsigned)t) >> 20;
    re[r] = xp[(size_t)t * CC];
    im[r] = 0.f;
  }
  __syncthreads();
  for (int lg = 1; lg <= 12; ++lg) {             // DIT stages
    int len = 1 << lg, half = len >> 1;
    float ang0 = -6.283185307179586f / (float)len;
    for (int u = tid; u < TT / 2; u += 256) {
      int off = u & (half - 1);
      int blk = u >> (lg - 1);
      int i0 = blk * len + off, i1 = i0 + half;
      float s, ct;
      __sincosf(ang0 * (float)off, &s, &ct);
      float br = re[i1], bi = im[i1];
      float tr = br * ct - bi * s;
      float ti = br * s + bi * ct;
      float ar = re[i0], ai = im[i0];
      re[i0] = ar + tr; im[i0] = ai + ti;
      re[i1] = ar - tr; im[i1] = ai - ti;
    }
    __syncthreads();
  }
  for (int f = 1 + tid; f <= FB; f += 256) {
    float m = sqrtf(re[f] * re[f] + im[f] * im[f]);
    mag[((size_t)b * FB + (f - 1)) * CC + c] = m;
  }
}

// ------------------------------------ mean over channels (fixed-order => det.)
__global__ void k_reduce(const float* __restrict__ mag, float* __restrict__ amps, int n) {
  int i = blockIdx.x * 256 + threadIdx.x;
  if (i >= n) return;
  const float* p = mag + (size_t)i * CC;
  float s = 0.f;
  for (int c = 0; c < CC; ++c) s += p[c];
  amps[i] = s * (1.0f / CC);
}

// ------------------------------------------- top-K + periods + softmax weights
__global__ __launch_bounds__(256) void k_topk(const float* __restrict__ amps,
                                              int* __restrict__ pk, float* __restrict__ wk) {
  __shared__ float vals[FB];
  __shared__ float rv[256];
  __shared__ int   ri[256];
  __shared__ float topv[KK];
  __shared__ int   topi[KK];
  int b = blockIdx.x, tid = threadIdx.x;
  for (int i = tid; i < FB; i += 256) vals[i] = amps[b * FB + i];
  __syncthreads();
  for (int k = 0; k < KK; ++k) {
    float best = -3.0e38f; int bi = 0;
    for (int i = tid; i < FB; i += 256)
      if (vals[i] > best) { best = vals[i]; bi = i; }
    rv[tid] = best; ri[tid] = bi;
    __syncthreads();
    for (int s = 128; s > 0; s >>= 1) {
      if (tid < s && rv[tid + s] > rv[tid]) { rv[tid] = rv[tid + s]; ri[tid] = ri[tid + s]; }
      __syncthreads();
    }
    if (tid == 0) { topv[k] = rv[0]; topi[k] = ri[0]; vals[ri[0]] = -3.0e38f; }
    __syncthreads();
  }
  if (tid == 0) {
    float m = topv[0];
    for (int k = 1; k < KK; ++k) m = fmaxf(m, topv[k]);
    float e[KK], s = 0.f;
    for (int k = 0; k < KK; ++k) { e[k] = __expf(topv[k] - m); s += e[k]; }
    for (int k = 0; k < KK; ++k) {
      int idx = topi[k];
      int denom = idx > 1 ? idx : 1;
      int p = TT / denom; if (p < 1) p = 1;
      pk[b * KK + k] = p;
      wk[b * KK + k] = e[k] / s;
    }
  }
}

// ------------------------------------------------------- conv as 9 shifted GEMMs
// fold/conv/unfold == out[t,:] += mask * x[t + dy*n + dx, :] @ Wc[dy,dx,:,:]
// LDS rows are zero-filled for t' outside [0,T), so only the fold-column
// condition needs masking (dx=-1: j>0 ; dx=+1: j<n-1) — done by selecting the
// LDS *address* between the real row and a dedicated zero row (1 VALU, not 8).
#define ZROW (3 * 66 * CC)   // offset (elements) of the 256-entry zero row
__global__ __launch_bounds__(256) void k_conv(const unsigned short* __restrict__ xbf,
                                              const unsigned short* __restrict__ wcf,
                                              const float* __restrict__ bc,
                                              const int* __restrict__ pk,
                                              unsigned short* __restrict__ tf) {
  __shared__ unsigned short ldsX[3 * 66 * CC + CC];   // 3 dy-blocks + zero row
  int t0 = blockIdx.x * 64;
  int bk = blockIdx.y;                            // b*K + k
  int b  = bk >> 2;
  int n  = pk[bk];
  int tid = threadIdx.x;
  int w = tid >> 5, lane = tid & 31;
  int hl = lane >> 4, ml = lane & 15;

  // cooperative stage via ASYNC global->LDS DMA (ASYNCcnt-tracked, bypasses VGPRs)
  // row r of dy-block holds x[t0 + r - 1 + dy*n]; OOB rows zero-filled with DS
  // stores at disjoint addresses (no ordering hazard with the async loads).
  for (int it = tid; it < 3 * 66 * 64; it += 256) {
    int dyb = it / (66 * 64);
    int rr  = it - dyb * (66 * 64);
    int r = rr >> 6, c4 = rr & 63;
    int tp = t0 + r - 1 + (dyb - 1) * n;
    unsigned short* lp = &ldsX[(dyb * 66 + r) * CC + c4 * 4];
    if (tp >= 0 && tp < TT) {
      unsigned ldsOff = (unsigned)(size_t)lp;      // LDS aperture: addr[31:0]
      unsigned long long gaddr =
          (unsigned long long)(size_t)&xbf[((size_t)b * TT + tp) * CC + c4 * 4];
      asm volatile("global_load_async_to_lds_b64 %0, %1, off"
                   :: "v"(ldsOff), "v"(gaddr) : "memory");
    } else {
      *(uint2*)lp = make_uint2(0u, 0u);
    }
  }
  for (int it = tid; it < 64; it += 256)          // zero row for masked A rows
    *(uint2*)&ldsX[ZROW + it * 4] = make_uint2(0u, 0u);
  asm volatile("s_wait_asynccnt 0x0" ::: "memory");
  __syncthreads();

  v8f z = {};
  v8f acc[4][2];
  for (int mt = 0; mt < 4; ++mt) { acc[mt][0] = z; acc[mt][1] = z; }
  int jrow[4];
  for (int mt = 0; mt < 4; ++mt) {
    int t = t0 + mt * 16 + ml;
    jrow[mt] = t - (t / n) * n;                   // t % n (runtime n)
  }

  for (int tap = 0; tap < 9; ++tap) {
    int dy = tap / 3 - 1, dx = tap % 3 - 1;
    // next tap's weight fragments -> global_prefetch_b8
    if (tap < 8)
      __builtin_prefetch(wcf + (size_t)(((tap + 1) * 8) * 16 + 2 * w) * 512 + lane * 32, 0, 1);
    unsigned vmask[4];
    for (int mt = 0; mt < 4; ++mt)
      vmask[mt] = (dx < 0) ? (jrow[mt] > 0) : ((dx > 0) ? (jrow[mt] < n - 1) : 1u);
#pragma unroll
    for (int kc = 0; kc < 8; ++kc) {
      Frag bfr[2];
      for (int j = 0; j < 2; ++j) {               // wave owns N cols [32w,32w+32)
        int tile = (tap * 8 + kc) * 16 + (2 * w + j);
        const uint4* p = (const uint4*)wcf + (size_t)(tile * 32 + lane) * 2;
        bfr[j].q[0] = p[0]; bfr[j].q[1] = p[1];
      }
      int zoff = ZROW + kc * 32 + 16 * hl;
      for (int mt = 0; mt < 4; ++mt) {
        int r = mt * 16 + ml + 1 + dx;
        int roff = ((dy + 1) * 66 + r) * CC + kc * 32 + 16 * hl;
        int off = vmask[mt] ? roff : zoff;        // address select, not data select
        const uint4* p = (const uint4*)&ldsX[off];
        Frag a;
        a.q[0] = p[0];
        a.q[1] = p[1];                            // next 8 contiguous K elems
        acc[mt][0] = __builtin_amdgcn_wmma_f32_16x16x32_bf16(
            false, a.v, false, bfr[0].v, (short)0, acc[mt][0], false, false);
        acc[mt][1] = __builtin_amdgcn_wmma_f32_16x16x32_bf16(
            false, a.v, false, bfr[1].v, (short)0, acc[mt][1], false, false);
      }
    }
  }
  // epilogue: +bias, exact GELU, store bf16 into tf[b,k,t,c]
  for (int j = 0; j < 2; ++j) {
    int c = 32 * w + j * 16 + ml;
    float bcv = bc[c];
    for (int mt = 0; mt < 4; ++mt)
      for (int r = 0; r < 8; ++r) {
        int t = t0 + mt * 16 + r + 8 * hl;
        float v = acc[mt][j][r] + bcv;
        tf[((size_t)bk * TT + t) * CC + c] = f2bf(gelu_exact(v));
      }
  }
}

// ----------------------------- fused FFN: GELU(tf@W1+b1)@W2, *softmax weight,
// written in place into tf (block-exclusive tile; barrier before writes)
__global__ __launch_bounds__(256) void k_ffn(unsigned short* __restrict__ tf,
                                             const unsigned short* __restrict__ w1f,
                                             const unsigned short* __restrict__ w2f,
                                             const float* __restrict__ b1,
                                             const float* __restrict__ wk) {
  __shared__ unsigned short hc_s[64 * 72];        // 64x64 hidden chunk, padded stride
  int t0 = blockIdx.x * 64;
  int bk = blockIdx.y;
  int tid = threadIdx.x;
  int w = tid >> 5, lane = tid & 31;
  int hl = lane >> 4, ml = lane & 15;
  int mt1 = w >> 1;                               // GEMM1: wave's M-tile
  int ntBase1 = 2 * (w & 1);                      // GEMM1: wave's two local N-tiles

  v8f z = {};
  v8f acc2[4][2];
  for (int mt = 0; mt < 4; ++mt) { acc2[mt][0] = z; acc2[mt][1] = z; }

  const size_t tfBase = ((size_t)bk * TT) * CC;

  // hoist GEMM1 A fragments (tf is read-only until the final epilogue)
  Frag aC[8];
  {
    int t = t0 + mt1 * 16 + ml;
    for (int kc = 0; kc < 8; ++kc) {
      const uint4* pa = (const uint4*)&tf[tfBase + (size_t)t * CC + kc * 32 + 16 * hl];
      aC[kc].q[0] = pa[0]; aC[kc].q[1] = pa[1];
    }
  }

  for (int hc = 0; hc < 16; ++hc) {               // hidden dim 1024 in chunks of 64
    v8f hacc[2]; hacc[0] = z; hacc[1] = z;
#pragma unroll
    for (int kc = 0; kc < 8; ++kc) {
      for (int j = 0; j < 2; ++j) {
        int tile = kc * 64 + hc * 4 + ntBase1 + j;
        const uint4* pb = (const uint4*)w1f + (size_t)(tile * 32 + lane) * 2;
        Frag bf; bf.q[0] = pb[0]; bf.q[1] = pb[1];
        hacc[j] = __builtin_amdgcn_wmma_f32_16x16x32_bf16(
            false, aC[kc].v, false, bf.v, (short)0, hacc[j], false, false);
      }
    }
    __syncthreads();                              // prev iter's hc_s reads done
    for (int j = 0; j < 2; ++j) {
      int ntL = ntBase1 + j;
      int ncol = ntL * 16 + ml;
      float b1v = b1[hc * 64 + ncol];
      for (int r = 0; r < 8; ++r) {
        int row = mt1 * 16 + r + 8 * hl;
        hc_s[row * 72 + ncol] = f2bf(gelu_exact(hacc[j][r] + b1v));
      }
    }
    __syncthreads();
    for (int kc2 = 0; kc2 < 2; ++kc2) {           // GEMM2: accumulate into acc2
      Frag bfr[2];
      for (int j = 0; j < 2; ++j) {
        int tile = (hc * 2 + kc2) * 16 + 2 * w + j;
        const uint4* pb = (const uint4*)w2f + (size_t)(tile * 32 + lane) * 2;
        bfr[j].q[0] = pb[0]; bfr[j].q[1] = pb[1];
      }
      for (int mt = 0; mt < 4; ++mt) {
        int row = mt * 16 + ml;
        const uint4* pa = (const uint4*)&hc_s[row * 72 + kc2 * 32 + 16 * hl];
        Frag a; a.q[0] = pa[0]; a.q[1] = pa[1];
        acc2[mt][0] = __builtin_amdgcn_wmma_f32_16x16x32_bf16(
            false, a.v, false, bfr[0].v, (short)0, acc2[mt][0], false, false);
        acc2[mt][1] = __builtin_amdgcn_wmma_f32_16x16x32_bf16(
            false, a.v, false, bfr[1].v, (short)0, acc2[mt][1], false, false);
      }
    }
  }
  __syncthreads();                                // all tf reads done before overwrite
  float wv = wk[bk];
  for (int j = 0; j < 2; ++j) {
    int c = 32 * w + j * 16 + ml;
    for (int mt = 0; mt < 4; ++mt)
      for (int r = 0; r < 8; ++r) {
        int t = t0 + mt * 16 + r + 8 * hl;
        tf[tfBase + (size_t)t * CC + c] = f2bf(acc2[mt][j][r] * wv);
      }
  }
}

// ----------------------- out = x + b2 + sum_k tf_k   (softmax weights sum to 1)
__global__ void k_final(const float* __restrict__ x, const float* __restrict__ b2,
                        const unsigned short* __restrict__ tf, float* __restrict__ out) {
  size_t i = (size_t)blockIdx.x * 256 + threadIdx.x;
  if (i >= (size_t)BB * TT * CC) return;
  int c = (int)(i % CC);
  size_t bt = i / CC;
  int b = (int)(bt / TT);
  size_t t = bt % TT;
  float s = x[i] + b2[c];
  for (int k = 0; k < KK; ++k)
    s += bf2f(tf[(((size_t)(b * KK + k) * TT) + t) * CC + c]);
  out[i] = s;
}

extern "C" void kernel_launch(void* const* d_in, const int* in_sizes, int n_in,
                              void* d_out, int out_size, void* d_ws, size_t ws_size,
                              hipStream_t stream) {
  const float* x  = (const float*)d_in[0];
  const float* Wc = (const float*)d_in[1];
  const float* bc = (const float*)d_in[2];
  const float* W1 = (const float*)d_in[3];
  const float* b1 = (const float*)d_in[4];
  const float* W2 = (const float*)d_in[5];
  const float* b2 = (const float*)d_in[6];
  float* out = (float*)d_out;

  // workspace carve-up (256B aligned)
  char* ws = (char*)d_ws;
  size_t off = 0;
  auto alloc = [&](size_t bytes) { char* p = ws + off; off += (bytes + 255) & ~(size_t)255; return p; };
  unsigned short* xbf  = (unsigned short*)alloc((size_t)BB * TT * CC * 2);       // 16 MB
  unsigned short* tf   = (unsigned short*)alloc((size_t)BB * KK * TT * CC * 2);  // 64 MB
  float*          mag  = (float*)alloc((size_t)BB * FB * CC * 4);                // 16 MB
  float*          amps = (float*)alloc((size_t)BB * FB * 4);
  unsigned short* wcf  = (unsigned short*)alloc((size_t)9 * 256 * 256 * 2);
  unsigned short* w1f  = (unsigned short*)alloc((size_t)256 * 1024 * 2);
  unsigned short* w2f  = (unsigned short*)alloc((size_t)1024 * 256 * 2);
  int*            pk   = (int*)alloc(BB * KK * 4);
  float*          wkp  = (float*)alloc(BB * KK * 4);

  const int BTC = BB * TT * CC;
  k_xbf<<<(BTC + 255) / 256, 256, 0, stream>>>(x, xbf, BTC);
  k_frags<<<(9 * 256 * 256 + 255) / 256, 256, 0, stream>>>(Wc, wcf, 256, 256, 9);
  k_frags<<<(256 * 1024 + 255) / 256, 256, 0, stream>>>(W1, w1f, 256, 1024, 1);
  k_frags<<<(1024 * 256 + 255) / 256, 256, 0, stream>>>(W2, w2f, 1024, 256, 1);
  k_fft<<<BB * CC, 256, 0, stream>>>(x, mag);
  k_reduce<<<(BB * FB + 255) / 256, 256, 0, stream>>>(mag, amps, BB * FB);
  k_topk<<<BB, 256, 0, stream>>>(amps, pk, wkp);
  k_conv<<<dim3(TT / 64, BB * KK), 256, 0, stream>>>(xbf, wcf, bc, pk, tf);
  k_ffn<<<dim3(TT / 64, BB * KK), 256, 0, stream>>>(tf, w1f, w2f, b1, wkp);
  k_final<<<(BTC + 255) / 256, 256, 0, stream>>>(x, b2, tf, out);
}
